// BayesianAttention_82076825026674
// MI455X (gfx1250) — matrix operand order repaired and checked by reading, not compiled, not run on hardware
//
#include <hip/hip_runtime.h>
#include <hip/hip_bf16.h>
#include <math.h>

// ---------------------------------------------------------------------------
// BayesianAttention for MI455X (gfx1250, wave32, WMMA 16x16x32 f16,
// async global->LDS staging of the Weibull noise stream)
// B=8 L=512 D=1024 H=16 DK=DV=64
// ---------------------------------------------------------------------------

typedef __attribute__((ext_vector_type(16))) _Float16 v16h;
typedef __attribute__((ext_vector_type(8)))  float    v8f;
typedef __attribute__((vector_size(16)))     int      v4i_vec;
typedef __attribute__((address_space(1))) v4i_vec* gptr_v4i;  // global int4*
typedef __attribute__((address_space(3))) v4i_vec* lptr_v4i;  // LDS    int4*

#define Bdim 8
#define Ldim 512
#define Ddim 1024
#define Hdim 16
#define DK   64
#define DV   64
#define EPSc 1e-20f
#define EULER_G 0.5772156649015329f

#if defined(__has_builtin)
#if __has_builtin(__builtin_amdgcn_global_load_async_to_lds_b128) && \
    __has_builtin(__builtin_amdgcn_s_wait_asynccnt)
#define USE_ASYNC_LDS 1
#endif
#endif
#ifndef USE_ASYNC_LDS
#define USE_ASYNC_LDS 0
#endif

// ---------------- WMMA fragment helpers (wave32) ---------------------------
// A-frag (16x32 f16): lanes 0-15 hold M=lane, K=0..15 contiguous;
//                     lanes 16-31 hold M=lane-16, K=16..31 contiguous.
__device__ __forceinline__ v16h load_a_f16(const _Float16* __restrict__ base,
                                           int row0, int row_stride) {
  int lane = threadIdx.x & 31;
  const _Float16* p = base + (size_t)(row0 + (lane & 15)) * row_stride
                           + ((lane < 16) ? 0 : 16);
  v16h a;
#pragma unroll
  for (int j = 0; j < 16; ++j) a[j] = p[j];
  return a;
}
// B-frag (32x16 f16): lane n(0-15): col n, K=0..15 contiguous;
//                     lane n(16-31): col n-16, K=16..31 contiguous.
// base must point at K-contiguous storage of B's columns (i.e. Bt rows).
__device__ __forceinline__ v16h load_b_f16(const _Float16* __restrict__ base,
                                           int col0, int col_stride) {
  int lane = threadIdx.x & 31;
  const _Float16* p = base + (size_t)(col0 + (lane & 15)) * col_stride
                           + ((lane < 16) ? 0 : 16);
  v16h b;
#pragma unroll
  for (int j = 0; j < 16; ++j) b[j] = p[j];
  return b;
}
__device__ __forceinline__ v8f wmma32(v16h a, v16h b, v8f c) {
  return __builtin_amdgcn_wmma_f32_16x16x32_f16(false, a, false, b,
                                                (short)0, c, false, false);
}

// ---------------- Generic 64x64-tile GEMM:  C = A@W + bias -----------------
// A: [M,K] (f32 or f16), W: [K,N] f32, C: [M,N] (f32 or f16). 256 thr, 8 waves,
// each wave owns two 16x16 output tiles; K staged through LDS in f16.
template <bool A_F16, bool OUT_F16>
__global__ __launch_bounds__(256) void gemm_wmma(
    const void* __restrict__ Aptr, const float* __restrict__ W,
    const float* __restrict__ bias, void* __restrict__ Cptr,
    int M, int N, int K) {
  __shared__ _Float16 Atile[64 * 32];
  __shared__ _Float16 Wt[64 * 32];  // W tile stored column-of-B contiguous in K
  const int t = threadIdx.x, lane = t & 31, wv = t >> 5;
  const int n0 = blockIdx.x * 64, m0 = blockIdx.y * 64;
  const int tt0 = wv * 2, tt1 = wv * 2 + 1;
  const int rt0 = tt0 >> 2, ct0 = tt0 & 3, rt1 = tt1 >> 2, ct1 = tt1 & 3;
  v8f acc0 = {}, acc1 = {};

  for (int k0 = 0; k0 < K; k0 += 32) {
    const int m = t >> 2, kk = (t & 3) * 8;
    if (A_F16) {
      const _Float16* A = (const _Float16*)Aptr;
#pragma unroll
      for (int j = 0; j < 8; ++j)
        Atile[m * 32 + kk + j] = A[(size_t)(m0 + m) * K + k0 + kk + j];
    } else {
      const float* A = (const float*)Aptr;
#pragma unroll
      for (int j = 0; j < 8; ++j)
        Atile[m * 32 + kk + j] = (_Float16)A[(size_t)(m0 + m) * K + k0 + kk + j];
    }
#pragma unroll
    for (int j = 0; j < 8; ++j)
      Wt[m * 32 + kk + j] = (_Float16)W[(size_t)(k0 + kk + j) * N + n0 + m];
    // prefetch next K-tile while this one is consumed (global_prefetch_b8)
    if (k0 + 32 < K) {
      if (A_F16)
        __builtin_prefetch((const _Float16*)Aptr + (size_t)(m0 + m) * K +
                               k0 + 32 + kk, 0, 1);
      else
        __builtin_prefetch((const float*)Aptr + (size_t)(m0 + m) * K +
                               k0 + 32 + kk, 0, 1);
      __builtin_prefetch(&W[(size_t)(k0 + 32 + kk) * N + n0 + m], 0, 1);
    }
    __syncthreads();
    acc0 = wmma32(load_a_f16(Atile, rt0 * 16, 32),
                  load_b_f16(Wt, ct0 * 16, 32), acc0);
    acc1 = wmma32(load_a_f16(Atile, rt1 * 16, 32),
                  load_b_f16(Wt, ct1 * 16, 32), acc1);
    __syncthreads();
  }
#pragma unroll
  for (int pass = 0; pass < 2; ++pass) {
    const v8f acc = pass ? acc1 : acc0;
    const int rt = pass ? rt1 : rt0, ct = pass ? ct1 : ct0;
    const int col = n0 + ct * 16 + (lane & 15);
    const int rb = m0 + rt * 16 + ((lane < 16) ? 0 : 8);
    const float bvv = bias[col];
#pragma unroll
    for (int i = 0; i < 8; ++i) {
      const float v = acc[i] + bvv;
      if (OUT_F16)
        ((_Float16*)Cptr)[(size_t)(rb + i) * N + col] = (_Float16)v;
      else
        ((float*)Cptr)[(size_t)(rb + i) * N + col] = v;
    }
  }
}

// ---------------- V transpose: [B*L, H*DV] f16 -> [B,H,DV,L] f16 -----------
__global__ __launch_bounds__(256) void transpose_v(
    const _Float16* __restrict__ v, _Float16* __restrict__ vT) {
  size_t idx = (size_t)blockIdx.x * 256 + threadIdx.x;
  if (idx >= (size_t)Bdim * Hdim * DV * Ldim) return;
  int l = idx & (Ldim - 1);
  size_t rest = idx >> 9;
  int dv = rest & (DV - 1);
  size_t bh = rest >> 6;
  int b = (int)(bh >> 4), h = (int)(bh & 15);
  vT[idx] = v[((size_t)(b * Ldim + l)) * (Hdim * DV) + h * DV + dv];
}

// ---------------- Prior MLP: dot_gamma[b,h,l] ------------------------------
__global__ __launch_bounds__(256) void prior_mlp(
    const _Float16* __restrict__ kf16, const float* __restrict__ Wp1,
    const float* __restrict__ bp1, const float* __restrict__ Wp2,
    const float* __restrict__ bp2, float* __restrict__ dg) {
  const int wg = blockIdx.x * 8 + (threadIdx.x >> 5);  // 0..B*H*L-1
  const int lane = threadIdx.x & 31;
  const int l = wg & (Ldim - 1);
  const int bh = wg >> 9, b = bh >> 4, h = bh & 15;
  const _Float16* krow = kf16 + ((size_t)(b * Ldim + l)) * (Hdim * DK) + h * DK;
  float kv[DK];
#pragma unroll
  for (int c = 0; c < DK; ++c) kv[c] = (float)krow[c];
  float sum = 0.f;
  for (int d = lane; d < Ddim; d += 32) {
    float a = bp1[d];
#pragma unroll 8
    for (int c = 0; c < DK; ++c) a = fmaf(kv[c], Wp1[c * Ddim + d], a);
    a = (a >= 0.f) ? a : 0.01f * a;           // LeakyReLU
    sum = fmaf(Wp2[d], a, sum);
  }
#pragma unroll
  for (int off = 16; off; off >>= 1) sum += __shfl_xor(sum, off, 32);
  if (lane == 0) dg[wg] = sum + bp2[0];
}

// ---------------- alpha = softmax_K(dot_gamma); accumulate KL S2 -----------
__global__ __launch_bounds__(256) void alpha_softmax(
    const float* __restrict__ dg, float* __restrict__ alpha,
    float* __restrict__ accum) {
  __shared__ float red[256];
  const int bh = blockIdx.x, t = threadIdx.x;
  const float* row = dg + (size_t)bh * Ldim;
  float v0 = row[t], v1 = row[t + 256];
  red[t] = fmaxf(v0, v1);
  __syncthreads();
  for (int s = 128; s; s >>= 1) {
    if (t < s) red[t] = fmaxf(red[t], red[t + s]);
    __syncthreads();
  }
  const float m = red[0];
  __syncthreads();
  const float e0 = expf(v0 - m), e1 = expf(v1 - m);
  red[t] = e0 + e1;
  __syncthreads();
  for (int s = 128; s; s >>= 1) {
    if (t < s) red[t] += red[t + s];
    __syncthreads();
  }
  const float invZ = 1.0f / red[0];
  __syncthreads();
  const float a0 = e0 * invZ, a1 = e1 * invZ;  // BETA_GAMMA == 1
  alpha[(size_t)bh * Ldim + t] = a0;
  alpha[(size_t)bh * Ldim + t + 256] = a1;
  // S2 per-key term: alpha*lgamma(3) + 2*gamma_E*alpha - alpha*ln(beta+eps)
  //                  + lgamma(alpha+eps)
  const float cterm = lgammaf(3.0f) + 2.0f * EULER_G - logf(1.0f + EPSc);
  red[t] = a0 * cterm + lgammaf(a0 + EPSc) + a1 * cterm + lgammaf(a1 + EPSc);
  __syncthreads();
  for (int s = 128; s; s >>= 1) {
    if (t < s) red[t] += red[t + s];
    __syncthreads();
  }
  if (t == 0) atomicAdd(accum + 1, red[0]);
}

// ---------------- Fused stochastic attention -------------------------------
// One block = 32 query rows of one (b,h). Scores kept entirely in 64KB LDS;
// the 64KB unif block for this tile is staged global->LDS asynchronously,
// overlapping the 134MB noise read with the score WMMAs.
__global__ __launch_bounds__(256) void attention(
    const _Float16* __restrict__ qf16, const _Float16* __restrict__ kf16,
    const _Float16* __restrict__ vT, const float* __restrict__ alpha,
    const float* __restrict__ unif, _Float16* __restrict__ opre,
    float* __restrict__ accum) {
  __shared__ float S[32 * Ldim];  // 64 KB
#if USE_ASYNC_LDS
  __shared__ float U[32 * Ldim];  // 64 KB staged noise
#endif
  const int bx = blockIdx.x;
  const int bh = bx >> 4, q0 = (bx & 15) * 32;
  const int b = bh >> 4, h = bh & 15;
  const int t = threadIdx.x, lane = t & 31, wv = t >> 5;

#if USE_ASYNC_LDS
  {
    // rows q0..q0+31 of unif[bh] are contiguous: 32*512 f32 = 64KB = 4096 x b128
    float* ug = (float*)unif + ((size_t)bh * Ldim + q0) * Ldim;
#pragma unroll
    for (int i = 0; i < 16; ++i) {
      const int c = t + 256 * i;  // 16B chunk index
      __builtin_amdgcn_global_load_async_to_lds_b128(
          (gptr_v4i)(ug + (size_t)c * 4),
          (lptr_v4i)((char*)U + (size_t)c * 16), 0, 0);
    }
  }
#endif

  // ---- phase 1: S = (Q @ K^T) * sqrt(DK), via WMMA, into LDS ----
  const _Float16* qbase = qf16 + ((size_t)(b * Ldim + q0)) * (Hdim * DK) + h * DK;
  const _Float16* kbase = kf16 + ((size_t)(b * Ldim)) * (Hdim * DK) + h * DK;
  for (int ti = 0; ti < 8; ++ti) {
    const int tt = wv * 8 + ti;
    const int rt = tt >> 5, kt = tt & 31;
    v8f acc = {};
#pragma unroll
    for (int c = 0; c < 2; ++c) {  // DK=64 -> two K=32 steps
      v16h a = load_a_f16(qbase + c * 32, rt * 16, Hdim * DK);
      v16h bb = load_b_f16(kbase + c * 32, kt * 16, Hdim * DK);
      acc = wmma32(a, bb, acc);
    }
    const int col = kt * 16 + (lane & 15);
    const int rb = rt * 16 + ((lane < 16) ? 0 : 8);
#pragma unroll
    for (int i = 0; i < 8; ++i) S[(rb + i) * Ldim + col] = acc[i] * 8.0f;
  }
#if USE_ASYNC_LDS
  __builtin_amdgcn_s_wait_asynccnt(0);  // noise block resident in LDS
#endif
  __syncthreads();

  // ---- phase 2: softmax -> logprobs -> Weibull perturb -> softmax, + KL ----
  {
    const int r = t >> 3;          // 8 consecutive lanes own one row
    const int c0 = (t & 7) * 64;   // each owns 64 of 512 columns
    float* Sr = S + (size_t)r * Ldim;
    float m1 = -1e30f;
    for (int j = 0; j < 64; ++j) m1 = fmaxf(m1, Sr[c0 + j]);
#pragma unroll
    for (int off = 1; off < 8; off <<= 1) m1 = fmaxf(m1, __shfl_xor(m1, off, 32));
    float Z1 = 0.f;
    for (int j = 0; j < 64; ++j) Z1 += expf(Sr[c0 + j] - m1);
#pragma unroll
    for (int off = 1; off < 8; off <<= 1) Z1 += __shfl_xor(Z1, off, 32);
    const float lZ1 = logf(Z1);
#if USE_ASYNC_LDS
    const float* urow = U + (size_t)r * Ldim;
#else
    const float* urow = unif + ((size_t)bh * Ldim + q0 + r) * Ldim;
#endif
    const float* arow = alpha + (size_t)bh * Ldim;
    float kl = 0.f, m2 = -1e30f;
    for (int j = 0; j < 64; ++j) {
      const int c = c0 + j;
      const float ls = Sr[c] - m1 - lZ1;
      const float p = expf(ls);
      const float lp = logf(p + EPSc);                 // log(softmax + eps)
      const float u = urow[c];
      const float g = logf(-logf(1.0f - u + EPSc) + EPSc);
      const float z = lp - lgammaf(3.0f + 2.0f * g);   // 1 + 1/k + g/k, k=0.5
      Sr[c] = z;
      kl += (p + EPSc) - arow[c] * lp;                 // S1 element
      m2 = fmaxf(m2, z);
    }
#pragma unroll
    for (int off = 1; off < 8; off <<= 1) m2 = fmaxf(m2, __shfl_xor(m2, off, 32));
    float Z2 = 0.f;
    for (int j = 0; j < 64; ++j) Z2 += expf(Sr[c0 + j] - m2);
#pragma unroll
    for (int off = 1; off < 8; off <<= 1) Z2 += __shfl_xor(Z2, off, 32);
    const float inv = 1.0f / Z2;
    for (int j = 0; j < 64; ++j) Sr[c0 + j] = expf(Sr[c0 + j] - m2) * inv;
#pragma unroll
    for (int off = 1; off < 32; off <<= 1) kl += __shfl_xor(kl, off, 32);
    if (lane == 0) atomicAdd(accum + 0, kl);
  }
  __syncthreads();

  // ---- phase 3: out = att_s @ V via WMMA (att from LDS f32 -> f16 frags) ----
  {
    const int rt = wv >> 2, ct = wv & 3;  // 2 row-tiles x 4 dv-tiles
    v8f acc = {};
    const float* Sb = S + (size_t)(rt * 16 + (lane & 15)) * Ldim
                        + ((lane < 16) ? 0 : 16);
    const _Float16* vb = vT + ((size_t)bh * DV + ct * 16 + (lane & 15)) * Ldim
                            + ((lane < 16) ? 0 : 16);
    for (int kk = 0; kk < 16; ++kk) {   // 512 keys = 16 x K32
      v16h a;
#pragma unroll
      for (int j = 0; j < 16; ++j) {
        a[j] = (_Float16)Sb[kk * 32 + j];
      }
      v16h bb;
      const _Float16* p = vb + kk * 32;
#pragma unroll
      for (int j = 0; j < 16; ++j) bb[j] = p[j];
      acc = wmma32(a, bb, acc);
    }
    const int col = h * DV + ct * 16 + (lane & 15);
    const int rb = q0 + rt * 16 + ((lane < 16) ? 0 : 8);
#pragma unroll
    for (int i = 0; i < 8; ++i)
      opre[((size_t)(b * Ldim + rb + i)) * (Hdim * DV) + col] = (_Float16)acc[i];
  }
}

// ---------------- tiny kernels --------------------------------------------
__global__ void zero_accum(float* accum) {
  if (threadIdx.x < 2) accum[threadIdx.x] = 0.f;
}
__global__ void kl_finalize(const float* __restrict__ accum,
                            float* __restrict__ out_kl) {
  if (threadIdx.x == 0) {
    const float n1 = (float)Bdim * Hdim * Ldim * Ldim;
    const float n2 = (float)Bdim * Hdim * Ldim;
    *out_kl = accum[0] / n1 + accum[1] / n2;
  }
}

// ---------------------------------------------------------------------------
extern "C" void kernel_launch(void* const* d_in, const int* in_sizes, int n_in,
                              void* d_out, int out_size, void* d_ws,
                              size_t ws_size, hipStream_t stream) {
  const float* queries = (const float*)d_in[0];
  const float* keys    = (const float*)d_in[1];
  const float* values  = (const float*)d_in[2];
  const float* unif    = (const float*)d_in[3];
  const float* Wq = (const float*)d_in[4];  const float* bq = (const float*)d_in[5];
  const float* Wk = (const float*)d_in[6];  const float* bk = (const float*)d_in[7];
  const float* Wv = (const float*)d_in[8];  const float* bv = (const float*)d_in[9];
  const float* Wo = (const float*)d_in[10]; const float* bo = (const float*)d_in[11];
  const float* Wp1 = (const float*)d_in[12]; const float* bp1 = (const float*)d_in[13];
  const float* Wp2 = (const float*)d_in[14]; const float* bp2 = (const float*)d_in[15];
  float* out = (float*)d_out;

  char* ws = (char*)d_ws;
  const size_t MB = 1u << 20;
  _Float16* qf16 = (_Float16*)(ws + 0 * MB);    // 4096x1024 f16 = 8MB
  _Float16* kf16 = (_Float16*)(ws + 8 * MB);
  _Float16* vf16 = (_Float16*)(ws + 16 * MB);
  _Float16* vT   = (_Float16*)(ws + 24 * MB);   // [B,H,DV,L] f16
  _Float16* opre = (_Float16*)(ws + 32 * MB);   // pre-Wo attention out f16
  float* dg      = (float*)(ws + 40 * MB);      // [B,H,L]
  float* alpha   = (float*)(ws + 40 * MB + (1u << 18));
  float* accum   = (float*)(ws + 40 * MB + (2u << 18));  // {S1, S2}

  zero_accum<<<1, 32, 0, stream>>>(accum);

  const dim3 gg(Ddim / 64, (Bdim * Ldim) / 64);  // (16, 64)
  gemm_wmma<false, true><<<gg, 256, 0, stream>>>(queries, Wq, bq, (void*)qf16,
                                                 Bdim * Ldim, Hdim * DK, Ddim);
  gemm_wmma<false, true><<<gg, 256, 0, stream>>>(keys, Wk, bk, (void*)kf16,
                                                 Bdim * Ldim, Hdim * DK, Ddim);
  gemm_wmma<false, true><<<gg, 256, 0, stream>>>(values, Wv, bv, (void*)vf16,
                                                 Bdim * Ldim, Hdim * DV, Ddim);

  const int nvt = Bdim * Hdim * DV * Ldim;
  transpose_v<<<(nvt + 255) / 256, 256, 0, stream>>>(vf16, vT);

  prior_mlp<<<(Bdim * Hdim * Ldim) / 8, 256, 0, stream>>>(kf16, Wp1, bp1, Wp2,
                                                          bp2, dg);
  alpha_softmax<<<Bdim * Hdim, 256, 0, stream>>>(dg, alpha, accum);

  attention<<<Bdim * Hdim * (Ldim / 32), 256, 0, stream>>>(
      qf16, kf16, vT, alpha, unif, opre, accum);

  gemm_wmma<true, false><<<gg, 256, 0, stream>>>(opre, Wo, bo, (void*)out,
                                                 Bdim * Ldim, Ddim, Hdim * DV);

  kl_finalize<<<1, 32, 0, stream>>>(accum, out + (size_t)Bdim * Ldim * Ddim);
}
